// DIFFormerConv_44487271252726
// MI455X (gfx1250) — matrix operand order repaired and checked by reading, not compiled, not run on hardware
//
#include <hip/hip_runtime.h>

// CDNA5 / gfx1250, wave32. WMMA f16 (16x16x32, f32 accum) everywhere.
// Fragment layouts exploit the fact that with K stored innermost, each WMMA
// operand fragment is 1-2 contiguous 16B blocks per lane (CDNA5 ISA §7.12.2):
//   A 16x32: elems 0-7  <- K=8*hi..8*hi+7 ; elems 8-15 <- K=16+8*hi..16+8*hi+7
//   B 32x16: elems 0-7  <- K=8*hi..8*hi+7 ; elems 8-15 <- K=16+8*hi..16+8*hi+7
typedef __attribute__((ext_vector_type(16))) _Float16 v16h;
typedef __attribute__((ext_vector_type(8)))  float    v8f;

#define B_   4
#define N_   16384
#define DIN  256
#define DOUT 512
#define NH   8
#define DM   64
#define LSTR 40  // LDS row stride in halves (80B: keeps 16B alignment, bank-spread)

__device__ __forceinline__ v8f wmma_f16(v16h a, v16h b, v8f c) {
  // (neg_a, A, neg_b, B, c_mod, C, reuse_a, reuse_b)
  return __builtin_amdgcn_wmma_f32_16x16x32_f16(false, a, false, b, (short)0, c,
                                                false, false);
}

// A-matrix fragment: base points at [row][k=0]; two 16B blocks.
__device__ __forceinline__ v16h fragA(const _Float16* base, int hi) {
  union { v16h v; uint4 u[2]; } x;
  x.u[0] = *(const uint4*)(base + 8 * hi);
  x.u[1] = *(const uint4*)(base + 16 + 8 * hi);
  return x.v;
}
// B-matrix fragment: base points at [col][k=0]; two 16B blocks (same k order as A).
__device__ __forceinline__ v16h fragB(const _Float16* base, int hi) {
  union { v16h v; uint4 u[2]; } x;
  x.u[0] = *(const uint4*)(base + 8 * hi);
  x.u[1] = *(const uint4*)(base + 16 + 8 * hi);
  return x.v;
}

// Convert 8 fp32 -> f16 and store as one 16B chunk.
__device__ __forceinline__ void cvt_store8(_Float16* dst, float4 f0, float4 f1) {
  union { _Float16 h[8]; uint4 u; } x;
  x.h[0] = (_Float16)f0.x; x.h[1] = (_Float16)f0.y;
  x.h[2] = (_Float16)f0.z; x.h[3] = (_Float16)f0.w;
  x.h[4] = (_Float16)f1.x; x.h[5] = (_Float16)f1.y;
  x.h[6] = (_Float16)f1.z; x.h[7] = (_Float16)f1.w;
  *(uint4*)dst = x.u;
}

// ---------------------------------------------------------------------------
// Kernel A: fused QKV GEMM.  C[r, j] = X[r,:] . W[j,:] + bias[j]
// grid = (512 row-tiles of 128, 12 col-tiles of 128: 4xQ, 4xK, 4xV)
// Also accumulates global sum-of-squares of Q and K (Frobenius norms).
// ---------------------------------------------------------------------------
__global__ __launch_bounds__(256) void qkv_gemm_wmma(
    const float* __restrict__ Xq, const float* __restrict__ Xs,
    const float* __restrict__ Wq, const float* __restrict__ Wk,
    const float* __restrict__ Wv,
    const float* __restrict__ bq, const float* __restrict__ bk,
    const float* __restrict__ bv,
    _Float16* __restrict__ Qb, _Float16* __restrict__ Kb,
    _Float16* __restrict__ Vb, float* __restrict__ sumsq) {
  __shared__ _Float16 As[128][LSTR];  // activations [row][k]
  __shared__ _Float16 Bs[128][LSTR];  // weights     [col][k]  (W is row-major in k)
  __shared__ float    Red[256];

  const int t    = threadIdx.x;
  const int ct   = blockIdx.y;
  const int seg  = ct >> 2;           // 0=Q, 1=K, 2=V
  const int segc = (ct & 3) * 128;    // column base inside DOUT
  const int row0 = blockIdx.x * 128;  // flattened B*N row base

  const float* X    = (seg == 0) ? Xq : Xs;
  const float* W    = (seg == 0) ? Wq : (seg == 1) ? Wk : Wv;
  const float* bias = (seg == 0) ? bq : (seg == 1) ? bk : bv;
  _Float16*    Out  = (seg == 0) ? Qb : (seg == 1) ? Kb : Vb;

  const int wave = t >> 5, lane = t & 31;
  const int l = lane & 15, hi = lane >> 4;
  const int rowb = (wave >> 2) * 64;  // wave sub-tile: 64 rows x 32 cols
  const int colb = (wave & 3) * 32;

  const v8f vz = {0.f, 0.f, 0.f, 0.f, 0.f, 0.f, 0.f, 0.f};
  v8f acc[4][2];
#pragma unroll
  for (int i = 0; i < 4; ++i)
#pragma unroll
    for (int j = 0; j < 2; ++j) acc[i][j] = vz;

  const int ar = t >> 1, akk = (t & 1) * 16;  // loader: 16 elems/thread/tile

  for (int k0 = 0; k0 < DIN; k0 += 32) {
    const float* pa = X + (size_t)(row0 + ar) * DIN + k0 + akk;
    const float* pb = W + (size_t)(segc + ar) * DIN + k0 + akk;
    if (k0 + 32 < DIN) {  // global_prefetch_b8 for next k-chunk
      __builtin_prefetch(pa + 32, 0, 3);
      __builtin_prefetch(pb + 32, 0, 3);
    }
#pragma unroll
    for (int q = 0; q < 2; ++q) {
      cvt_store8(&As[ar][akk + q * 8], *(const float4*)(pa + q * 8),
                 *(const float4*)(pa + q * 8 + 4));
      cvt_store8(&Bs[ar][akk + q * 8], *(const float4*)(pb + q * 8),
                 *(const float4*)(pb + q * 8 + 4));
    }
    __syncthreads();

    v16h afrag[4], bfrag[2];
#pragma unroll
    for (int ri = 0; ri < 4; ++ri) afrag[ri] = fragA(&As[rowb + ri * 16 + l][0], hi);
#pragma unroll
    for (int ci = 0; ci < 2; ++ci) bfrag[ci] = fragB(&Bs[colb + ci * 16 + l][0], hi);
#pragma unroll
    for (int ri = 0; ri < 4; ++ri)
#pragma unroll
      for (int ci = 0; ci < 2; ++ci)
        acc[ri][ci] = wmma_f16(afrag[ri], bfrag[ci], acc[ri][ci]);
    __syncthreads();
  }

  // Epilogue: bias, f16 store, sum-of-squares (Q and K segments only).
  float ssq = 0.f;
#pragma unroll
  for (int ci = 0; ci < 2; ++ci) {
    const int gcol = segc + colb + ci * 16 + l;
    const float bvl = bias[gcol];
#pragma unroll
    for (int ri = 0; ri < 4; ++ri) {
#pragma unroll
      for (int i = 0; i < 8; ++i) {
        // C layout: lane<16 -> N=lane, M=vgpr; lane>=16 -> M=8+vgpr
        const int grow = row0 + rowb + ri * 16 + hi * 8 + i;
        const float v = acc[ri][ci][i] + bvl;
        Out[(size_t)grow * DOUT + gcol] = (_Float16)v;
        ssq += v * v;
      }
    }
  }
  if (seg < 2) {
    Red[t] = ssq;
    __syncthreads();
    for (int s = 128; s > 0; s >>= 1) {
      if (t < s) Red[t] += Red[t + s];
      __syncthreads();
    }
    if (t == 0) atomicAdd(&sumsq[seg], Red[0]);
  }
}

// ---------------------------------------------------------------------------
// Kernel C: kvs[b,h] = K[b,:,h]^T . V[b,:,h]  (64x64, reduce over N=16384)
// One workgroup per (b,h). LDS tiles stored K-innermost (transposed scatter
// at fill time) so all fragments are contiguous b128 loads.
// Emits kvs TRANSPOSED as kvsT[bh][d][m] plus ks_sum / vs_sum column sums.
// ---------------------------------------------------------------------------
__global__ __launch_bounds__(256) void kv_outer_wmma(
    const _Float16* __restrict__ Kb, const _Float16* __restrict__ Vb,
    _Float16* __restrict__ kvsT, float* __restrict__ ks_sum,
    float* __restrict__ vs_sum) {
  __shared__ _Float16 KsT[64][LSTR];  // [m][kr]
  __shared__ _Float16 VsT[64][LSTR];  // [d][kr]
  __shared__ float    Pk[32][64];
  __shared__ float    Pv[32][64];

  const int t  = threadIdx.x;
  const int bh = blockIdx.x;
  const int bb = bh >> 3, h = bh & 7;
  const int wave = t >> 5, lane = t & 31;
  const int l = lane & 15, hi = lane >> 4;
  const int mt  = wave >> 1;        // each wave: 1 m-tile x 2 d-tiles
  const int dt0 = (wave & 1) * 2;

  const int r = t >> 3, c8 = (t & 7) * 8;  // loader: 8 f16 = 16B per thread

  const v8f vz = {0.f, 0.f, 0.f, 0.f, 0.f, 0.f, 0.f, 0.f};
  v8f acc0 = vz, acc1 = vz;
  float pk[8], pv[8];
#pragma unroll
  for (int i = 0; i < 8; ++i) { pk[i] = 0.f; pv[i] = 0.f; }

  for (int n0 = 0; n0 < N_; n0 += 32) {
    const size_t goff = (size_t)(bb * N_ + n0 + r) * DOUT + h * DM + c8;
    const uint4 uk = *(const uint4*)(Kb + goff);
    const uint4 uv = *(const uint4*)(Vb + goff);
    const _Float16* hk = (const _Float16*)&uk;
    const _Float16* hv = (const _Float16*)&uv;
#pragma unroll
    for (int i = 0; i < 8; ++i) {      // transposed scatter into LDS
      KsT[c8 + i][r] = hk[i];
      VsT[c8 + i][r] = hv[i];
      pk[i] += (float)hk[i];
      pv[i] += (float)hv[i];
    }
    __syncthreads();

    const v16h a   = fragA(&KsT[mt * 16 + l][0], hi);          // A = K^T
    const v16h bf0 = fragB(&VsT[dt0 * 16 + l][0], hi);
    const v16h bf1 = fragB(&VsT[(dt0 + 1) * 16 + l][0], hi);
    acc0 = wmma_f16(a, bf0, acc0);
    acc1 = wmma_f16(a, bf1, acc1);
    __syncthreads();
  }

  // Column-sum reduction (ks_sum / vs_sum).
#pragma unroll
  for (int i = 0; i < 8; ++i) {
    Pk[r][c8 + i] = pk[i];
    Pv[r][c8 + i] = pv[i];
  }
  __syncthreads();
  if (t < 64) {
    float sk = 0.f, sv = 0.f;
    for (int rg = 0; rg < 32; ++rg) {
      sk += Pk[rg][t];
      sv += Pv[rg][t];
    }
    ks_sum[bh * DM + t] = sk;
    vs_sum[bh * DM + t] = sv;
  }

  // Store kvs transposed ([d][m]); each lane's 8 values are contiguous in m.
#pragma unroll
  for (int ci = 0; ci < 2; ++ci) {
    const int d  = (dt0 + ci) * 16 + l;
    const int m0 = mt * 16 + hi * 8;
    const v8f av = ci ? acc1 : acc0;
    union { _Float16 h[8]; uint4 u; } x;
#pragma unroll
    for (int i = 0; i < 8; ++i) x.h[i] = (_Float16)av[i];
    *(uint4*)&kvsT[((size_t)bh * DM + d) * DM + m0] = x.u;
  }
}

__global__ void init_scal(float* scal) {
  if (threadIdx.x < 4) scal[threadIdx.x] = 0.f;
}
__global__ void finish_scale(float* scal) {
  if (threadIdx.x == 0) scal[2] = rsqrtf(scal[0] * scal[1]);  // 1/(||Q||*||K||)
}

// ---------------------------------------------------------------------------
// Kernel E: den[b,n,h] = q[b,n,h,:] . ks_sum[b,h,:]
// ---------------------------------------------------------------------------
__global__ __launch_bounds__(256) void qdotk_den(
    const _Float16* __restrict__ Qb, const float* __restrict__ ks_sum,
    float* __restrict__ den) {
  const int idx = blockIdx.x * 256 + threadIdx.x;  // exactly B*N*H threads
  const int h = idx & 7;
  const size_t grow = (size_t)(idx >> 3);
  const int bb = (int)(grow >> 14);
  const _Float16* q = Qb + grow * DOUT + h * DM;
  const float* ks = ks_sum + (size_t)((bb << 3) + h) * DM;
  float a = 0.f;
#pragma unroll
  for (int c = 0; c < 8; ++c) {
    const uint4 u = ((const uint4*)q)[c];
    const _Float16* hp = (const _Float16*)&u;
#pragma unroll
    for (int i = 0; i < 8; ++i) a += (float)hp[i] * ks[c * 8 + i];
  }
  den[idx] = a;
}

// ---------------------------------------------------------------------------
// Kernel D: out[b,n,:] = mean_h (s*(q.kvs) + vs_sum) / (s*den + N)
// 64 rows per block; wave = (row-tile, head-group); head-mean via LDS atomics.
// B-frags come from the transposed kvsT so every fragment is contiguous.
// ---------------------------------------------------------------------------
__global__ __launch_bounds__(256) void attn_final_wmma(
    const _Float16* __restrict__ Qb, const _Float16* __restrict__ kvsT,
    const float* __restrict__ vs_sum, const float* __restrict__ den,
    const float* __restrict__ scal, float* __restrict__ out) {
  __shared__ float Racc[64][64];
  const int t = threadIdx.x;
  const size_t row0 = (size_t)blockIdx.x * 64;
  const int bb = (int)(row0 >> 14);
  const int wave = t >> 5, lane = t & 31;
  const int l = lane & 15, hi = lane >> 4;
  const int rt = wave & 3, hg = wave >> 2;
  const float s = scal[2];

  for (int i = t; i < 64 * 64; i += 256) ((float*)Racc)[i] = 0.f;
  __syncthreads();

  const v8f vz = {0.f, 0.f, 0.f, 0.f, 0.f, 0.f, 0.f, 0.f};
  for (int hh = 0; hh < 4; ++hh) {
    const int h = hg * 4 + hh;
    const int bh = (bb << 3) + h;
    v8f acc[4];
#pragma unroll
    for (int dt = 0; dt < 4; ++dt) acc[dt] = vz;

#pragma unroll
    for (int ks0 = 0; ks0 < DM; ks0 += 32) {
      const size_t grow = row0 + rt * 16 + l;
      const v16h a = fragA(Qb + grow * DOUT + h * DM + ks0, hi);
#pragma unroll
      for (int dt = 0; dt < 4; ++dt) {
        const int d = dt * 16 + l;
        // kvsT[bh][d][m]: B[k=m][n=d] pairs contiguous in m.
        const v16h bf = fragB(kvsT + ((size_t)bh * DM + d) * DM + ks0, hi);
        acc[dt] = wmma_f16(a, bf, acc[dt]);
      }
    }

    float invdn[8];
#pragma unroll
    for (int i = 0; i < 8; ++i) {
      const size_t grow = row0 + rt * 16 + hi * 8 + i;
      invdn[i] = 1.0f / (s * den[grow * NH + h] + (float)N_);
    }
#pragma unroll
    for (int dt = 0; dt < 4; ++dt) {
      const int d = dt * 16 + l;
      const float vsum = vs_sum[bh * DM + d];
#pragma unroll
      for (int i = 0; i < 8; ++i) {
        const int rbi = rt * 16 + hi * 8 + i;
        atomicAdd(&Racc[rbi][d], (s * acc[dt][i] + vsum) * invdn[i]);
      }
    }
  }
  __syncthreads();
  for (int i = t; i < 64 * 64; i += 256) {
    const int rbi = i >> 6, d = i & 63;
    out[(row0 + rbi) * DM + d] = Racc[rbi][d] * 0.125f;  // mean over 8 heads
  }
}

// ---------------------------------------------------------------------------
extern "C" void kernel_launch(void* const* d_in, const int* in_sizes, int n_in,
                              void* d_out, int out_size, void* d_ws,
                              size_t ws_size, hipStream_t stream) {
  const float* Xq = (const float*)d_in[0];
  const float* Xs = (const float*)d_in[1];
  const float* Wq = (const float*)d_in[2];
  const float* bq = (const float*)d_in[3];
  const float* Wk = (const float*)d_in[4];
  const float* bk = (const float*)d_in[5];
  const float* Wv = (const float*)d_in[6];
  const float* bv = (const float*)d_in[7];
  float* out = (float*)d_out;

  char* ws = (char*)d_ws;
  size_t o = 0;
  float* scal = (float*)(ws + o); o += 256;                  // [ssqQ, ssqK, s, -]
  const size_t qkvBytes = (size_t)B_ * N_ * DOUT * sizeof(_Float16);
  _Float16* Qb = (_Float16*)(ws + o); o += qkvBytes;         // 64 MB
  _Float16* Kb = (_Float16*)(ws + o); o += qkvBytes;         // 64 MB
  _Float16* Vb = (_Float16*)(ws + o); o += qkvBytes;         // 64 MB
  _Float16* kvsT = (_Float16*)(ws + o);
  o += (size_t)B_ * NH * DM * DM * sizeof(_Float16);         // 256 KB
  float* ks_sum = (float*)(ws + o); o += (size_t)B_ * NH * DM * 4;
  float* vs_sum = (float*)(ws + o); o += (size_t)B_ * NH * DM * 4;
  float* den = (float*)(ws + o); o += (size_t)B_ * N_ * NH * 4;  // 2 MB

  hipLaunchKernelGGL(init_scal, dim3(1), dim3(32), 0, stream, scal);
  hipLaunchKernelGGL(qkv_gemm_wmma, dim3((B_ * N_) / 128, 12), dim3(256), 0,
                     stream, Xq, Xs, Wq, Wk, Wv, bq, bk, bv, Qb, Kb, Vb, scal);
  hipLaunchKernelGGL(kv_outer_wmma, dim3(B_ * NH), dim3(256), 0, stream, Kb, Vb,
                     kvsT, ks_sum, vs_sum);
  hipLaunchKernelGGL(finish_scale, dim3(1), dim3(32), 0, stream, scal);
  hipLaunchKernelGGL(qdotk_den, dim3((B_ * N_ * NH) / 256), dim3(256), 0,
                     stream, Qb, ks_sum, den);
  hipLaunchKernelGGL(attn_final_wmma, dim3((B_ * N_) / 64), dim3(256), 0,
                     stream, Qb, kvsT, vs_sum, den, scal, out);
}
